// CellularAutomata_48627619725659
// MI455X (gfx1250) — matrix operand order repaired and checked by reading, not compiled, not run on hardware
//
#include <hip/hip_runtime.h>

typedef __attribute__((ext_vector_type(16))) _Float16 v16h;
typedef __attribute__((ext_vector_type(8)))  float    v8f;

#define BATCH 8
#define SZ    256
#define CH    16
#define HID   128
#define KPAD  64          // K=48 zero-padded to 64 (two 16x16x32 WMMA K-blocks)
#define WAVES 4
#define TPB   (WAVES * 32)

union AFrag { v16h v; unsigned u[8]; };

__global__ __launch_bounds__(TPB)
void nca_wmma_kernel(const float* __restrict__ x,
                     const float* __restrict__ pk,   // (3,3,3) [ky][kx][t]
                     const float* __restrict__ W1,   // (48,128)
                     const float* __restrict__ b1,   // (128)
                     const float* __restrict__ W2,   // (128,16)
                     const float* __restrict__ b2,   // (16)
                     float* __restrict__ out)
{
    // ---- LDS ----
    __shared__ __align__(32) _Float16 w1t[HID][KPAD];        // W1 transposed [n][k], k>=48 zero
    __shared__ __align__(32) _Float16 w2t[CH][HID];          // W2 transposed [n][k]
    __shared__ float                  pker[32];              // 27 perception coeffs
    __shared__ __align__(32) _Float16 s_tile[WAVES][16][KPAD];  // perception A-panel
    __shared__ __align__(32) _Float16 h_tile[WAVES][16][HID];   // hidden A-panel
    __shared__ __align__(16) float    x_tile[WAVES][16][CH];    // center x for residual

    const int tid  = threadIdx.x;
    const int w    = tid >> 5;
    const int lane = tid & 31;
    const int lo   = lane & 15;   // pixel (A rows / D cols index)
    const int hi   = lane >> 4;   // half-wave select

    // ---- stage weights (transposed, f32->f16) ----
    for (int i = tid; i < HID * KPAD; i += TPB) {
        int n = i / KPAD, k = i % KPAD;
        w1t[n][k] = (_Float16)(k < 48 ? W1[k * HID + n] : 0.f);
    }
    for (int i = tid; i < CH * HID; i += TPB) {
        int n = i / HID, k = i % HID;
        w2t[n][k] = (_Float16)W2[k * CH + n];
    }
    if (tid < 27) pker[tid] = pk[tid];
    __syncthreads();

    // ---- tile coordinates: 16 consecutive pixels in one row per wave ----
    const int tile = blockIdx.x * WAVES + w;          // 32768 tiles total
    const int xblk = tile & 15;                       // 16 x-blocks per row
    const int y    = (tile >> 4) & (SZ - 1);
    const int bi   = tile >> 12;
    const int xc   = (xblk << 4) + lo;                // this lane's pixel x
    const size_t imgbase = (size_t)bi * SZ * SZ * CH;

    // ---- perception: lane handles pixel `lo`, channels 8*hi .. 8*hi+7 ----
    const int c0 = hi * 8;
    for (int q = 0; q < 2; ++q) {
        const int cbase = c0 + q * 4;
        float4 nb[9];
        #pragma unroll
        for (int j = 0; j < 9; ++j) {
            int yy = (y + j / 3 - 1) & (SZ - 1);
            int xx = (xc + j % 3 - 1) & (SZ - 1);
            nb[j] = *(const float4*)(x + imgbase + ((size_t)yy * SZ + xx) * CH + cbase);
        }
        *(float4*)&x_tile[w][lo][cbase] = nb[4];      // center, for residual
        const float* fn = reinterpret_cast<const float*>(nb);
        #pragma unroll
        for (int e = 0; e < 4; ++e) {
            #pragma unroll
            for (int t = 0; t < 3; ++t) {
                float acc = 0.f;
                #pragma unroll
                for (int j = 0; j < 9; ++j) acc += fn[j * 4 + e] * pker[j * 3 + t];
                s_tile[w][lo][(cbase + e) * 3 + t] = (_Float16)acc;
            }
        }
    }
    // zero-pad K = 48..63 (each lane pads 8 halves of its pixel row)
    #pragma unroll
    for (int i = 0; i < 8; ++i) s_tile[w][lo][48 + hi * 8 + i] = (_Float16)0.f;

    // ---- load A fragments of s (contiguous K order: element s holds K = s + 16*(lane>>4), same as B) ----
    AFrag a0, a1;
    {
        const _Float16* row = &s_tile[w][lo][0];
        #pragma unroll
        for (int i = 0; i < 8; ++i) {
            int K = 2 * i + 16 * hi;
            a0.u[i] = *(const unsigned*)(row + K);
            a1.u[i] = *(const unsigned*)(row + 32 + K);
        }
    }

    // ---- GEMM1: h = relu(s @ W1 + b1), 8 N-tiles x 2 K-blocks ----
    #pragma unroll
    for (int nt = 0; nt < 8; ++nt) {
        v8f acc = {};
        const int ncol = nt * 16 + lo;
        v16h bf0 = *(const v16h*)&w1t[ncol][16 * hi];
        v16h bf1 = *(const v16h*)&w1t[ncol][32 + 16 * hi];
        acc = __builtin_amdgcn_wmma_f32_16x16x32_f16(false, a0.v, false, bf0,
                                                     (short)0, acc, false, false);
        acc = __builtin_amdgcn_wmma_f32_16x16x32_f16(false, a1.v, false, bf1,
                                                     (short)0, acc, false, false);
        const float bias = b1[ncol];
        #pragma unroll
        for (int r = 0; r < 8; ++r) {
            float v = acc[r] + bias;
            v = v > 0.f ? v : 0.f;
            h_tile[w][r + 8 * hi][ncol] = (_Float16)v;   // D layout: M=r+8*hi, N=ncol
        }
    }

    // ---- GEMM2: dx = h @ W2, 4 K-blocks ----
    v8f acc2 = {};
    #pragma unroll
    for (int kb = 0; kb < 4; ++kb) {
        AFrag a2;
        const _Float16* hrow = &h_tile[w][lo][0];
        #pragma unroll
        for (int i = 0; i < 8; ++i) {
            int K = kb * 32 + 2 * i + 16 * hi;
            a2.u[i] = *(const unsigned*)(hrow + K);
        }
        v16h bf = *(const v16h*)&w2t[lo][kb * 32 + 16 * hi];
        acc2 = __builtin_amdgcn_wmma_f32_16x16x32_f16(false, a2.v, false, bf,
                                                      (short)0, acc2, false, false);
    }

    // ---- epilogue: residual + clamp, store ----
    const float b2v = b2[lo];
    #pragma unroll
    for (int r = 0; r < 8; ++r) {
        const int p = r + 8 * hi;                       // pixel within tile
        float o = x_tile[w][p][lo] + acc2[r] + b2v;
        o = fminf(fmaxf(o, 0.f), 1.f);
        out[imgbase + ((size_t)y * SZ + (xblk * 16 + p)) * CH + lo] = o;
    }
}

extern "C" void kernel_launch(void* const* d_in, const int* in_sizes, int n_in,
                              void* d_out, int out_size, void* d_ws, size_t ws_size,
                              hipStream_t stream) {
    (void)in_sizes; (void)n_in; (void)d_ws; (void)ws_size; (void)out_size;
    const float* x  = (const float*)d_in[0];
    const float* pk = (const float*)d_in[1];
    const float* W1 = (const float*)d_in[2];
    const float* b1 = (const float*)d_in[3];
    const float* W2 = (const float*)d_in[4];
    const float* b2 = (const float*)d_in[5];
    float* out = (float*)d_out;

    const int tiles  = BATCH * SZ * SZ / 16;   // 32768
    const int blocks = tiles / WAVES;          // 8192
    nca_wmma_kernel<<<blocks, TPB, 0, stream>>>(x, pk, W1, b1, W2, b2, out);
}
